// MixtralSparseMoeBlock_84679575208320
// MI455X (gfx1250) — matrix operand order repaired and checked by reading, not compiled, not run on hardware
//
#include <hip/hip_runtime.h>

// MoE block (Mixtral-style) for gfx1250: router + dense-expert SwiGLU FFN
// using v_wmma_f32_16x16x32_bf16 with f32 accumulation.
// Grid = (T/64 token tiles) x (8 experts) = 256 blocks -> fills the WGP array.
// Token tile = 64 (4 WMMA row tiles) so every weight (B) fragment feeds 4
// WMMAs -> ~4 global loads + 2 bf16 converts per WMMA.
// Routing weight folded into hidden activations; expert contributions are
// summed into a zeroed output via global_atomic_add_f32.

#define HD 1024   // hidden size
#define MD 3584   // intermediate size
#define NE 8      // experts
#define TT 64     // tokens per block
#define NR 4      // WMMA row tiles per block (TT/16)
#define MC 128    // m-chunk per iteration (8 col tiles of 16 -> one per wave)

typedef __bf16 bf16;
typedef bf16  bf16x16 __attribute__((ext_vector_type(16)));
typedef float floatx8 __attribute__((ext_vector_type(8)));

__device__ __forceinline__ bf16 f2bf(float f) { return (bf16)f; }

#define ZERO8 (floatx8){0.f, 0.f, 0.f, 0.f, 0.f, 0.f, 0.f, 0.f}

// ---------------------------------------------------------------------------
// Kernel 0: zero the output accumulator region (deterministic start state)
// ---------------------------------------------------------------------------
__global__ void zero_kernel(float* __restrict__ p, int n) {
  int i = blockIdx.x * blockDim.x + threadIdx.x;
  if (i < n) p[i] = 0.f;
}

// ---------------------------------------------------------------------------
// Kernel 1: router logits + softmax + top-2 + renormalized routing weights
// ---------------------------------------------------------------------------
__global__ void router_kernel(const float* __restrict__ x,    // [T, H]
                              const float* __restrict__ gw,   // [H, E]
                              float* __restrict__ logits,     // [T, E] (d_out tail)
                              float* __restrict__ rw,         // [T, E] (ws)
                              int T) {
  int t = blockIdx.x * blockDim.x + threadIdx.x;
  if (t >= T) return;

  float acc[NE];
#pragma unroll
  for (int e = 0; e < NE; ++e) acc[e] = 0.f;

  const float* xp = x + (size_t)t * HD;
  for (int h = 0; h < HD; ++h) {
    float xv = xp[h];
    const float* g = gw + (size_t)h * NE;
#pragma unroll
    for (int e = 0; e < NE; ++e) acc[e] = fmaf(xv, g[e], acc[e]);
  }

#pragma unroll
  for (int e = 0; e < NE; ++e) logits[(size_t)t * NE + e] = acc[e];

  // softmax over 8
  float mx = acc[0];
#pragma unroll
  for (int e = 1; e < NE; ++e) mx = fmaxf(mx, acc[e]);
  float p[NE], s = 0.f;
#pragma unroll
  for (int e = 0; e < NE; ++e) { p[e] = __expf(acc[e] - mx); s += p[e]; }
  float inv = 1.f / s;
#pragma unroll
  for (int e = 0; e < NE; ++e) p[e] *= inv;

  // top-2
  int i0 = 0;
#pragma unroll
  for (int e = 1; e < NE; ++e) if (p[e] > p[i0]) i0 = e;
  int i1 = (i0 == 0) ? 1 : 0;
#pragma unroll
  for (int e = 0; e < NE; ++e) if (e != i0 && p[e] > p[i1]) i1 = e;

  float denom = p[i0] + p[i1];
#pragma unroll
  for (int e = 0; e < NE; ++e) {
    float w = 0.f;
    if (e == i0) w = p[i0] / denom;
    else if (e == i1) w = p[i1] / denom;
    rw[(size_t)t * NE + e] = w;
  }
}

// ---------------------------------------------------------------------------
// Kernel 2: one (expert, 64-token tile) per block.
//
// Fragment layouts used by this kernel (wave32; contiguous K, A and B alike):
//   A (16x32 bf16): lane L -> row L%16; VGPR v holds K pair
//       k0 = (L/16)*16 + v*2
//   B (32x16 bf16): lane L -> col L%16; element j -> K = (L/16)*16 + j
//   C/D (16x16 f32): lane L -> col L%16; VGPR v -> row v + 8*(L/16)
// ---------------------------------------------------------------------------
__launch_bounds__(256, 1)
__global__ void moe_kernel(const float* __restrict__ x,      // [T, H]
                           const float* __restrict__ wgate,  // [E, H, M]
                           const float* __restrict__ wup,    // [E, H, M]
                           const float* __restrict__ wdown,  // [E, M, H]
                           const float* __restrict__ rw,     // [T, E]
                           float* __restrict__ out,          // [T, H] (atomic acc)
                           int T) {
  __shared__ bf16  Xs[TT][HD + 8];   // 64 x 1032 bf16 = 132 KB (bank padded)
  __shared__ bf16  Hs[TT][MC + 8];   // 64 x 136  bf16 = 17.4 KB
  __shared__ float rws[TT];

  const int tid   = threadIdx.x;
  const int lane  = tid & 31;
  const int wave  = tid >> 5;        // 0..7
  const int lhalf = lane >> 4;       // 0 | 1
  const int lmod  = lane & 15;
  const int t0    = blockIdx.x * TT;
  const int e     = blockIdx.y;      // expert id

  // Stage X tile fp32 -> bf16 into LDS (coalesced); routing weights too
  for (int i = tid; i < TT * HD; i += 256) {
    int r = i >> 10, c = i & (HD - 1);
    Xs[r][c] = f2bf(x[(size_t)(t0 + r) * HD + c]);
  }
  if (tid < TT) rws[tid] = rw[(size_t)(t0 + tid) * NE + e];
  __syncthreads();

  const float* Wg = wgate + (size_t)e * HD * MD;
  const float* Wu = wup   + (size_t)e * HD * MD;
  const float* Wd = wdown + (size_t)e * MD * HD;

  // Down accumulators: [4 row tiles][8 col tiles] over H cols [128*wave, +128)
  floatx8 accD[NR * 8];
#pragma unroll
  for (int j = 0; j < NR * 8; ++j) accD[j] = ZERO8;

  for (int m0 = 0; m0 < MD; m0 += MC) {
    // ---- gate & up projections: this wave owns cols [m0+16*wave, +16)
    const int mcol = m0 + 16 * wave + lmod;       // B-operand column (this lane)
    floatx8 accG[NR] = {ZERO8, ZERO8, ZERO8, ZERO8};
    floatx8 accU[NR] = {ZERO8, ZERO8, ZERO8, ZERO8};

    for (int k0 = 0; k0 < HD; k0 += 32) {
      bf16x16 a[NR];                              // A frags, row tiles 0..3
#pragma unroll
      for (int r = 0; r < NR; ++r) {
#pragma unroll
        for (int v = 0; v < 8; ++v) {
          int kk = k0 + lhalf * 16 + 2 * v;
          a[r][2 * v]     = Xs[16 * r + lmod][kk];
          a[r][2 * v + 1] = Xs[16 * r + lmod][kk + 1];
        }
      }
      bf16x16 bg, bu;                             // B frags, shared by 4 rows
#pragma unroll
      for (int j = 0; j < 16; ++j) {
        int kk = k0 + lhalf * 16 + j;
        size_t off = (size_t)kk * MD + mcol;
        bg[j] = f2bf(Wg[off]);
        bu[j] = f2bf(Wu[off]);
      }
      if (k0 + 32 < HD) {   // global_prefetch_b8 hint for the next K slab
        __builtin_prefetch(&Wg[(size_t)(k0 + 32 + lhalf * 16) * MD + mcol], 0, 1);
        __builtin_prefetch(&Wu[(size_t)(k0 + 32 + lhalf * 16) * MD + mcol], 0, 1);
      }
#pragma unroll
      for (int r = 0; r < NR; ++r) {
        accG[r] = __builtin_amdgcn_wmma_f32_16x16x32_bf16(
            false, a[r], false, bg, (short)0, accG[r], false, false);
        accU[r] = __builtin_amdgcn_wmma_f32_16x16x32_bf16(
            false, a[r], false, bu, (short)0, accU[r], false, false);
      }
    }

    // ---- h = routing_weight * silu(g) * u  ->  LDS (bf16)
    __syncthreads();   // previous m-chunk's Hs fully consumed
#pragma unroll
    for (int r = 0; r < NR; ++r) {
#pragma unroll
      for (int v = 0; v < 8; ++v) {
        int row = 16 * r + v + 8 * lhalf;         // token row within tile
        float g  = accG[r][v];
        float u  = accU[r][v];
        float sg = g / (1.f + __expf(-g));        // silu
        Hs[row][16 * wave + lmod] = f2bf(rws[row] * sg * u);
      }
    }
    __syncthreads();

    // ---- down projection: accD += Hs[64 x 128] @ Wd[m0:m0+128, wave cols]
    for (int kc = 0; kc < MC; kc += 32) {
      bf16x16 a[NR];
#pragma unroll
      for (int r = 0; r < NR; ++r) {
#pragma unroll
        for (int v = 0; v < 8; ++v) {
          int kk = kc + lhalf * 16 + 2 * v;
          a[r][2 * v]     = Hs[16 * r + lmod][kk];
          a[r][2 * v + 1] = Hs[16 * r + lmod][kk + 1];
        }
      }
#pragma unroll
      for (int j = 0; j < 8; ++j) {
        const int ncol = 128 * wave + 16 * j + lmod;
        bf16x16 bd;                               // shared by 4 row tiles
#pragma unroll
        for (int jj = 0; jj < 16; ++jj) {
          int kk = m0 + kc + lhalf * 16 + jj;
          bd[jj] = f2bf(Wd[(size_t)kk * HD + ncol]);
        }
#pragma unroll
        for (int r = 0; r < NR; ++r) {
          accD[r * 8 + j] = __builtin_amdgcn_wmma_f32_16x16x32_bf16(
              false, a[r], false, bd, (short)0, accD[r * 8 + j], false, false);
        }
      }
    }
  }

  // ---- accumulate this expert's routed contribution into out (f32 atomics)
#pragma unroll
  for (int r = 0; r < NR; ++r) {
#pragma unroll
    for (int j = 0; j < 8; ++j) {
      int col = 128 * wave + 16 * j + lmod;
#pragma unroll
      for (int v = 0; v < 8; ++v) {
        int tok = t0 + 16 * r + v + 8 * lhalf;
        atomicAdd(&out[(size_t)tok * HD + col], accD[r * 8 + j][v]);
      }
    }
  }
}

// ---------------------------------------------------------------------------
extern "C" void kernel_launch(void* const* d_in, const int* in_sizes, int n_in,
                              void* d_out, int out_size, void* d_ws, size_t ws_size,
                              hipStream_t stream) {
  const float* x  = (const float*)d_in[0];   // hidden_states [2,1024,H]
  const float* gw = (const float*)d_in[1];   // gate_w [H,E]
  const float* wg = (const float*)d_in[2];   // w_gate [E,H,M]
  const float* wu = (const float*)d_in[3];   // w_up   [E,H,M]
  const float* wd = (const float*)d_in[4];   // w_down [E,M,H]

  const int T = in_sizes[0] / HD;            // 2048 tokens

  float* out    = (float*)d_out;             // [T,H] then router_logits [T,E]
  float* logits = out + (size_t)T * HD;
  float* rw     = (float*)d_ws;              // [T,E] routing weights (64 KB)

  const int n = T * HD;
  zero_kernel<<<(n + 255) / 256, 256, 0, stream>>>(out, n);
  router_kernel<<<(T + 255) / 256, 256, 0, stream>>>(x, gw, logits, rw, T);

  dim3 grid(T / TT, NE);
  moe_kernel<<<grid, 256, 0, stream>>>(x, wg, wu, wd, rw, out, T);
}
